// GroupedQueryAttention_231928234336
// MI455X (gfx1250) — matrix operand order repaired and checked by reading, not compiled, not run on hardware
//
#include <hip/hip_runtime.h>
#include <hip/hip_bf16.h>

#define S_LEN   4096
#define DMODEL  1024
#define NHEAD   16
#define NKV     4
#define HD      64

typedef __attribute__((ext_vector_type(16))) __bf16         v16bf;
typedef __attribute__((ext_vector_type(8)))  float          v8f;

struct Frag32B { uint4 lo, hi; };   // 32 bytes, bit-cast to v16bf

// ---------------- CDNA5 async global->LDS staging ----------------
#if defined(__has_builtin)
#if __has_builtin(__builtin_amdgcn_global_load_async_to_lds_b128)
#define HAVE_ASYNC_LDS 1
#endif
#endif

#ifdef HAVE_ASYNC_LDS
typedef int v4i __attribute__((vector_size(16)));
typedef __attribute__((address_space(1))) v4i gv4i;   // global (device) AS
typedef __attribute__((address_space(3))) v4i lv4i;   // LDS AS

__device__ __forceinline__ void async_copy16B(const unsigned short* g, unsigned short* l) {
    // one b128 per lane: LDS[l] = MEM[g], tracked by ASYNCcnt
    __builtin_amdgcn_global_load_async_to_lds_b128(
        (gv4i*)(void*)g, (lv4i*)(void*)l, 0, 0);
}
__device__ __forceinline__ void wait_async_all() {
#if __has_builtin(__builtin_amdgcn_s_wait_asynccnt)
    __builtin_amdgcn_s_wait_asynccnt(0);
#else
    asm volatile("s_wait_asynccnt 0x0" ::: "memory");
#endif
}
#endif

// ---------------- helpers ----------------

__device__ __forceinline__ unsigned short f2bf(float f) {
    unsigned u = __float_as_uint(f);
    u += 0x7FFFu + ((u >> 16) & 1u);   // round-to-nearest-even
    return (unsigned short)(u >> 16);
}

// A fragment: 16x32 bf16 from row-major source (M x K), leading dim lda.
// Contiguous-k layout used here (B uses the same; not the ISA table order): lane holds row M=lane&15; K = half*16 + h
//  -> two contiguous 8-halfword chunks at +0 and +8 elements.
__device__ __forceinline__ v16bf load_a16x32(const unsigned short* A, int lda, int lane) {
    const unsigned short* row = A + (size_t)(lane & 15) * lda + ((lane >> 4) & 1) * 16;
    Frag32B t;
    t.lo = *reinterpret_cast<const uint4*>(row);
    t.hi = *reinterpret_cast<const uint4*>(row + 8);
    return __builtin_bit_cast(v16bf, t);
}

// B fragment: 32x16 bf16 where the source is K-contiguous per output column.
// p points at this lane's 16 consecutive K elements (lane = col N, K = 16*half + h).
__device__ __forceinline__ v16bf load_b_kcontig(const unsigned short* p) {
    Frag32B t;
    t.lo = *reinterpret_cast<const uint4*>(p);
    t.hi = *reinterpret_cast<const uint4*>(p + 8);
    return __builtin_bit_cast(v16bf, t);
}

// ---------------- stage 0: fp32 -> bf16 (plain + transposing) ----------------

__global__ void f32_to_bf16_kernel(const float* __restrict__ in,
                                   unsigned short* __restrict__ out, int n) {
    int i = blockIdx.x * blockDim.x + threadIdx.x;
    int stride = gridDim.x * blockDim.x;
    for (; i < n; i += stride) out[i] = f2bf(in[i]);
}

// in: K x N (row-major f32) -> out: N x K (row-major bf16)
__global__ void transpose_f32_to_bf16_kernel(const float* __restrict__ in,
                                             unsigned short* __restrict__ out,
                                             int K, int N) {
    int i = blockIdx.x * blockDim.x + threadIdx.x;
    int stride = gridDim.x * blockDim.x;
    int total = K * N;
    for (; i < total; i += stride) {
        int k = i / N, n = i - k * N;
        out[(size_t)n * K + k] = f2bf(in[i]);
    }
}

// ---------------- stage 1: fused QKV projection + RoPE ----------------
// Weights are pre-transposed: Wt[N][K=DMODEL].
// grid: (S/64, NHEAD + 2*NKV). block: 128 threads (4 waves).
__global__ __launch_bounds__(128)
void qkv_rope_kernel(const unsigned short* __restrict__ xb,
                     const unsigned short* __restrict__ wqt,
                     const unsigned short* __restrict__ wkt,
                     const unsigned short* __restrict__ wvt,
                     unsigned short* __restrict__ qh,   // [H][S][HD], pre-scaled 1/8
                     unsigned short* __restrict__ kh,   // [KV][S][HD]
                     unsigned short* __restrict__ vt) { // [KV][HD][S] (transposed)
    const int lane = threadIdx.x & 31;
    const int wave = threadIdx.x >> 5;
    const int m0   = blockIdx.x * 64 + wave * 16;
    const int what = blockIdx.y;
    const int half = (lane >> 4) & 1;

    const unsigned short* Wt;
    int head;
    if (what < NHEAD)            { Wt = wqt; head = what; }
    else if (what < NHEAD + NKV) { Wt = wkt; head = what - NHEAD; }
    else                         { Wt = wvt; head = what - NHEAD - NKV; }
    const int n0 = head * HD;

    v8f acc[4] = {};
#pragma unroll 2
    for (int k0 = 0; k0 < DMODEL; k0 += 32) {
        v16bf a = load_a16x32(xb + (size_t)m0 * DMODEL + k0, DMODEL, lane);
#pragma unroll
        for (int f = 0; f < 4; ++f) {
            v16bf b = load_b_kcontig(
                Wt + (size_t)(n0 + f * 16 + (lane & 15)) * DMODEL + k0 + half * 16);
            acc[f] = __builtin_amdgcn_wmma_f32_16x16x32_bf16(
                false, a, false, b, (short)0, acc[f], false, false);
        }
    }

    if (what >= NHEAD + NKV) {
        // V: store transposed [kv][d][s] so P*V B-fragments are K-contiguous
        unsigned short* outv = vt + (size_t)head * HD * S_LEN;
#pragma unroll
        for (int f = 0; f < 4; ++f)
#pragma unroll
            for (int r = 0; r < 8; ++r)
                outv[(size_t)(f * 16 + (lane & 15)) * S_LEN + (m0 + r + half * 8)]
                    = f2bf(acc[f][r]);
        return;
    }

    // RoPE: pair (d, d+32) lives in fragments (0,2)/(1,3), same lane, same r.
    // For Q, also fold in the attention scale 1/sqrt(HD) = 0.125 (RoPE is linear).
    const float qscale = (what < NHEAD) ? 0.125f : 1.0f;
#pragma unroll
    for (int fp = 0; fp < 2; ++fp) {
        float dIdx = (float)(fp * 16 + (lane & 15));
        float invf = __expf(dIdx * -0.28782313662425575f);  // ln(1e4)/32
#pragma unroll
        for (int r = 0; r < 8; ++r) {
            float pos = (float)(m0 + r + half * 8);
            float sn, cs;
            __sincosf(pos * invf, &sn, &cs);
            float t1 = acc[fp][r], t2 = acc[fp + 2][r];
            acc[fp][r]     = (t1 * cs - t2 * sn) * qscale;
            acc[fp + 2][r] = (t1 * sn + t2 * cs) * qscale;
        }
    }

    unsigned short* out = (what < NHEAD ? qh : kh) + (size_t)head * S_LEN * HD;
#pragma unroll
    for (int f = 0; f < 4; ++f)
#pragma unroll
        for (int r = 0; r < 8; ++r)
            out[(size_t)(m0 + r + half * 8) * HD + f * 16 + (lane & 15)] = f2bf(acc[f][r]);
}

// ---------------- stage 2: flash attention ----------------
// Double-buffered async-DMA K/V tiles in LDS; 4 waves share each tile.
// grid: (S/64, NHEAD). block: 128 threads (4 waves); wave owns 16 query rows.
__global__ __launch_bounds__(128)
void flash_attn_kernel(const unsigned short* __restrict__ qh,
                       const unsigned short* __restrict__ kh,   // [KV][S][HD]
                       const unsigned short* __restrict__ vt,   // [KV][HD][S]
                       unsigned short* __restrict__ ah) {       // [S][DMODEL]
    __shared__ unsigned short kTile[2][32 * HD];    // 2 x 4 KB: keys x dims
    __shared__ unsigned short vTile[2][HD * 32];    // 2 x 4 KB: dims x keys
    __shared__ unsigned short ldsP[4][16 * 32];     // 4 KB: per-wave P staging

    const int tid  = threadIdx.x;                   // 0..127
    const int lane = tid & 31;
    const int wave = tid >> 5;
    const int h    = blockIdx.y;
    const int kv   = h >> 2;                        // N_GROUP = 4
    const int qm0  = blockIdx.x * 64 + wave * 16;
    const int half = (lane >> 4) & 1;

    const unsigned short* Q  = qh + ((size_t)h  * S_LEN + qm0) * HD;
    const unsigned short* Kb = kh + (size_t)kv * S_LEN * HD;
    const unsigned short* Vt = vt + (size_t)kv * HD * S_LEN;

    // stage one K/V key-block (32 keys) into LDS buffer `b`
    auto stage = [&](int b, int kb) {
        const unsigned short* gk = Kb + (size_t)kb * HD;   // contiguous 4 KB
        const int c1 = tid + 128;                          // 16B-chunk ids
#ifdef HAVE_ASYNC_LDS
        async_copy16B(gk + (size_t)tid * 8, kTile[b] + (size_t)tid * 8);
        async_copy16B(gk + (size_t)c1 * 8,  kTile[b] + (size_t)c1 * 8);
        async_copy16B(Vt + (size_t)(tid >> 2) * S_LEN + kb + (tid & 3) * 8,
                      vTile[b] + (size_t)tid * 8);
        async_copy16B(Vt + (size_t)(c1 >> 2) * S_LEN + kb + (c1 & 3) * 8,
                      vTile[b] + (size_t)c1 * 8);
#else
        uint4* lk = reinterpret_cast<uint4*>(kTile[b]);
        const uint4* g4 = reinterpret_cast<const uint4*>(gk);
        lk[tid] = g4[tid];
        lk[c1]  = g4[c1];
        uint4* lv = reinterpret_cast<uint4*>(vTile[b]);
        lv[tid] = *reinterpret_cast<const uint4*>(
            Vt + (size_t)(tid >> 2) * S_LEN + kb + (tid & 3) * 8);
        lv[c1]  = *reinterpret_cast<const uint4*>(
            Vt + (size_t)(c1 >> 2) * S_LEN + kb + (c1 & 3) * 8);
#endif
    };

    const v16bf qf0 = load_a16x32(Q, HD, lane);       // dims 0..31 (pre-scaled)
    const v16bf qf1 = load_a16x32(Q + 32, HD, lane);  // dims 32..63

    v8f of[4] = {};
    float m_i[8], l_i[8];
#pragma unroll
    for (int r = 0; r < 8; ++r) { m_i[r] = -1e30f; l_i[r] = 0.f; }

    stage(0, 0);                                      // preload first tile
    int buf = 0;

#pragma unroll 1
    for (int kb = 0; kb < S_LEN; kb += 32) {
#ifdef HAVE_ASYNC_LDS
        wait_async_all();        // this wave's staged loads have landed in LDS
#endif
        __syncthreads();         // all waves' loads landed; prev reads finished

        // issue async staging of the NEXT tile; overlaps with compute below
        if (kb + 32 < S_LEN) stage(buf ^ 1, kb + 32);

        const unsigned short* kt = kTile[buf];
        const unsigned short* vtl = vTile[buf];

        // ---- scores: S(16x32) = Q(16x64) x K^T(64x32) ----
        v8f sf[2] = {};
#pragma unroll
        for (int ks = 0; ks < 64; ks += 32) {
            v16bf a = (ks == 0) ? qf0 : qf1;
#pragma unroll
            for (int f = 0; f < 2; ++f) {
                // B[k][n] = K[kb + f*16 + n][ks + k]: contiguous in LDS row
                v16bf b = load_b_kcontig(
                    kt + (size_t)(f * 16 + (lane & 15)) * HD + ks + half * 16);
                sf[f] = __builtin_amdgcn_wmma_f32_16x16x32_bf16(
                    false, a, false, b, (short)0, sf[f], false, false);
            }
        }

        // ---- online softmax (rows live in 16-lane groups of the C layout) ----
#pragma unroll
        for (int r = 0; r < 8; ++r) {
            float s0 = sf[0][r];               // scale pre-folded into Q
            float s1 = sf[1][r];
            float rm = fmaxf(s0, s1);
#pragma unroll
            for (int mask = 1; mask < 16; mask <<= 1)
                rm = fmaxf(rm, __shfl_xor(rm, mask));
            float mn    = fmaxf(m_i[r], rm);
            float alpha = __expf(m_i[r] - mn);
            m_i[r] = mn;
            float p0 = __expf(s0 - mn);
            float p1 = __expf(s1 - mn);
            sf[0][r] = p0; sf[1][r] = p1;
            float rs = p0 + p1;
#pragma unroll
            for (int mask = 1; mask < 16; mask <<= 1)
                rs += __shfl_xor(rs, mask);
            l_i[r] = l_i[r] * alpha + rs;
#pragma unroll
            for (int f = 0; f < 4; ++f) of[f][r] *= alpha;
        }

        // ---- P: C-layout -> A-layout via per-wave LDS tile ----
        unsigned short* pw = ldsP[wave];
#pragma unroll
        for (int f = 0; f < 2; ++f)
#pragma unroll
            for (int r = 0; r < 8; ++r)
                pw[(r + half * 8) * 32 + f * 16 + (lane & 15)] = f2bf(sf[f][r]);
        v16bf pa = load_a16x32(pw, 32, lane);     // ds_load_b128 x2

        // ---- O += P(16x32) x V(32x64), V tile is dim-major in LDS ----
#pragma unroll
        for (int f = 0; f < 4; ++f) {
            // B[k][n] = V[kb + k][f*16 + n] = vTile[f*16 + n][k]
            v16bf b = load_b_kcontig(
                vtl + (size_t)(f * 16 + (lane & 15)) * 32 + half * 16);
            of[f] = __builtin_amdgcn_wmma_f32_16x16x32_bf16(
                false, pa, false, b, (short)0, of[f], false, false);
        }

        buf ^= 1;
    }

    // ---- finalize: O /= l, store bf16 to attn_out[S][DMODEL] ----
#pragma unroll
    for (int r = 0; r < 8; ++r) {
        float inv = 1.0f / l_i[r];
        size_t row = (size_t)(qm0 + r + half * 8) * DMODEL + h * HD;
#pragma unroll
        for (int f = 0; f < 4; ++f)
            ah[row + f * 16 + (lane & 15)] = f2bf(of[f][r] * inv);
    }
}

// ---------------- stage 3: output projection (bf16 x bf16^T -> f32) ----------------
// B is pre-transposed: Bt[N][K]. grid: (N/64, M/64). block: 128 threads.
__global__ __launch_bounds__(128)
void gemm_bf16t_f32_kernel(const unsigned short* __restrict__ A,
                           const unsigned short* __restrict__ Bt,
                           float* __restrict__ C, int M, int N, int K) {
    const int lane = threadIdx.x & 31;
    const int wave = threadIdx.x >> 5;
    const int m0   = blockIdx.y * 64 + wave * 16;
    const int n0   = blockIdx.x * 64;
    const int half = (lane >> 4) & 1;

    v8f acc[4] = {};
#pragma unroll 2
    for (int k0 = 0; k0 < K; k0 += 32) {
        v16bf a = load_a16x32(A + (size_t)m0 * K + k0, K, lane);
#pragma unroll
        for (int f = 0; f < 4; ++f) {
            v16bf b = load_b_kcontig(
                Bt + (size_t)(n0 + f * 16 + (lane & 15)) * K + k0 + half * 16);
            acc[f] = __builtin_amdgcn_wmma_f32_16x16x32_bf16(
                false, a, false, b, (short)0, acc[f], false, false);
        }
    }

#pragma unroll
    for (int f = 0; f < 4; ++f)
#pragma unroll
        for (int r = 0; r < 8; ++r)
            C[(size_t)(m0 + r + half * 8) * N + n0 + f * 16 + (lane & 15)] = acc[f][r];
}

// ---------------- host ----------------

extern "C" void kernel_launch(void* const* d_in, const int* in_sizes, int n_in,
                              void* d_out, int out_size, void* d_ws, size_t ws_size,
                              hipStream_t stream) {
    const float* x  = (const float*)d_in[0];
    const float* wq = (const float*)d_in[1];
    const float* wk = (const float*)d_in[2];
    const float* wv = (const float*)d_in[3];
    const float* wo = (const float*)d_in[4];
    float* out = (float*)d_out;

    const size_t n_x  = (size_t)S_LEN * DMODEL;
    const size_t n_wq = (size_t)DMODEL * NHEAD * HD;
    const size_t n_wk = (size_t)DMODEL * NKV * HD;
    const size_t n_wv = n_wk;
    const size_t n_wo = (size_t)DMODEL * DMODEL;
    const size_t n_q  = (size_t)NHEAD * S_LEN * HD;
    const size_t n_k  = (size_t)NKV * S_LEN * HD;
    const size_t n_v  = n_k;
    const size_t n_a  = (size_t)S_LEN * DMODEL;

    unsigned short* p   = (unsigned short*)d_ws;
    unsigned short* xb  = p;              p += n_x;
    unsigned short* wqt = p;              p += n_wq;   // [1024][1024] transposed
    unsigned short* wkt = p;              p += n_wk;   // [256][1024]  transposed
    unsigned short* wvt = p;              p += n_wv;   // [256][1024]  transposed
    unsigned short* wot = p;              p += n_wo;   // [1024][1024] transposed
    unsigned short* qh  = p;              p += n_q;    // [H][S][HD]
    unsigned short* kh  = p;              p += n_k;    // [KV][S][HD]
    unsigned short* vt  = p;              p += n_v;    // [KV][HD][S]
    unsigned short* ah  = p;              p += n_a;    // [S][DMODEL]

    // stage 0: convert (weights transposed so B-fragments are K-contiguous)
    f32_to_bf16_kernel<<<1024, 256, 0, stream>>>(x, xb, (int)n_x);
    transpose_f32_to_bf16_kernel<<<1024, 256, 0, stream>>>(wq, wqt, DMODEL, NHEAD * HD);
    transpose_f32_to_bf16_kernel<<<256,  256, 0, stream>>>(wk, wkt, DMODEL, NKV * HD);
    transpose_f32_to_bf16_kernel<<<256,  256, 0, stream>>>(wv, wvt, DMODEL, NKV * HD);
    transpose_f32_to_bf16_kernel<<<1024, 256, 0, stream>>>(wo, wot, DMODEL, DMODEL);

    // stage 1: fused QKV GEMM + RoPE (Q pre-scaled by 1/8, V stored transposed)
    qkv_rope_kernel<<<dim3(S_LEN / 64, NHEAD + 2 * NKV), 128, 0, stream>>>(
        xb, wqt, wkt, wvt, qh, kh, vt);

    // stage 2: flash attention with double-buffered async LDS tiles
    flash_attn_kernel<<<dim3(S_LEN / 64, NHEAD), 128, 0, stream>>>(qh, kh, vt, ah);

    // stage 3: output projection into f32 d_out
    gemm_bf16t_f32_kernel<<<dim3(DMODEL / 64, S_LEN / 64), 128, 0, stream>>>(
        ah, wot, out, S_LEN, DMODEL, DMODEL);
}
